// DynamicGraphNetwork_58909771432787
// MI455X (gfx1250) — matrix operand order repaired and checked by reading, not compiled, not run on hardware
//
#include <hip/hip_runtime.h>
#include <hip/hip_bf16.h>
#include <math.h>

#define T_   8
#define N_   2048
#define D_   64
#define EP_  96      // enhanced dim 80 padded to 3*32
#define H_   128
#define NH_  4
#define HD_  32
#define F_   256
#define OUT_ 64

typedef __attribute__((ext_vector_type(16))) _Float16 v16h;
typedef __attribute__((ext_vector_type(8)))  float    v8f;

// ---------- WMMA operand loaders (wave32 layouts per CDNA5 ISA 7.12.2) ----------

// A-matrix 16x32 f16, row-major source, row stride lda (halves).
// lane: row = lane&15 ; khalf = (lane>>4)*8 ; VGPR j<4: K=khalf+2j ; j>=4: K=16+khalf+2(j-4)
__device__ __forceinline__ v16h load_a_tile(const _Float16* A, int lda) {
  int lane  = threadIdx.x & 31;
  int row   = lane & 15;
  int khalf = (lane >> 4) << 3;
  const _Float16* p = A + row * lda;
  union { unsigned u[8]; v16h v; } r;
#pragma unroll
  for (int j = 0; j < 4; ++j) r.u[j]     = *(const unsigned*)(p + khalf + 2 * j);
#pragma unroll
  for (int j = 0; j < 4; ++j) r.u[4 + j] = *(const unsigned*)(p + 16 + khalf + 2 * j);
  return r.v;
}

// B-matrix 32x16 f16 from column-major storage Bt[col][k], stride ldb (halves).
// lane: col = lane&15 ; kbase = (lane>>4)*8 ; VGPR j<4: K=kbase+2j ; j>=4: K=16+kbase+2(j-4)
__device__ __forceinline__ v16h load_b_tile(const _Float16* Bt, int ldb) {
  int lane  = threadIdx.x & 31;
  int col   = lane & 15;
  int kbase = (lane >> 4) << 3;
  const _Float16* p = Bt + col * ldb + kbase;
  union { unsigned u[8]; v16h v; } r;
#pragma unroll
  for (int j = 0; j < 8; ++j) r.u[j] = *(const unsigned*)(p + ((j >> 2) << 4) + 2 * (j & 3));
  return r.v;
}

__device__ __forceinline__ v8f wmma_f16(v16h a, v16h b, v8f c) {
  return __builtin_amdgcn_wmma_f32_16x16x32_f16(false, a, false, b, (short)0, c, false, false);
}

__device__ __forceinline__ float gelu_tanh(float x) {
  float x3 = x * x * x;
  return 0.5f * x * (1.0f + tanhf(0.7978845608028654f * (x + 0.044715f * x3)));
}

// ---------- prep: enhanced features [T,N,96] f16 ----------
__global__ void prep_enh_kernel(const float* __restrict__ nodes, const float* __restrict__ ts,
                                const float* __restrict__ tf, _Float16* __restrict__ enh) {
  int idx = blockIdx.x * blockDim.x + threadIdx.x;   // t*N + n
  if (idx >= T_ * N_) return;
  int t = idx / N_;
  const float* nrow = nodes + (size_t)idx * D_;
  _Float16* erow = enh + (size_t)idx * EP_;
  float tsv = ts[idx];
#pragma unroll 8
  for (int i = 0; i < D_; ++i) erow[i] = (_Float16)nrow[i];
#pragma unroll
  for (int j = 0; j < 8; ++j) {
    float ang = tsv * tf[t * 8 + j];
    erow[64 + j] = (_Float16)__sinf(ang);
    erow[72 + j] = (_Float16)__cosf(ang);
  }
#pragma unroll
  for (int i = 80; i < EP_; ++i) erow[i] = (_Float16)0.f;
}

// ---------- transpose+pad f32 [B][K][C] -> f16 [B][C][Kpad] ----------
__global__ void transpose_pad_kernel(const float* __restrict__ src, _Float16* __restrict__ dst,
                                     int B, int K, int C, int Kpad) {
  long total = (long)B * C * Kpad;
  for (long i = blockIdx.x * (long)blockDim.x + threadIdx.x; i < total;
       i += (long)gridDim.x * blockDim.x) {
    int k = (int)(i % Kpad);
    long rem = i / Kpad;
    int c = (int)(rem % C);
    int b = (int)(rem / C);
    float v = (k < K) ? src[((long)b * K + k) * C + c] : 0.f;
    dst[i] = (_Float16)v;
  }
}

// ---------- QKV projections: per-wave 16x16 tile, K=96 (3 WMMA) ----------
__global__ void qkv_kernel(const _Float16* __restrict__ enh,
                           const _Float16* __restrict__ WqT, const _Float16* __restrict__ WkT,
                           const _Float16* __restrict__ WvT,
                           const float* __restrict__ bq, const float* __restrict__ bk,
                           const float* __restrict__ bv,
                           _Float16* __restrict__ q16, _Float16* __restrict__ k16,
                           _Float16* __restrict__ vT) {
  int wave = (blockIdx.x * blockDim.x + threadIdx.x) >> 5;
  int ct = wave & 7;  wave >>= 3;
  int nt = wave & 127; wave >>= 7;
  int mat = wave % 3;
  int t = wave / 3;
  if (t >= T_) return;
  int lane = threadIdx.x & 31;
  int col  = lane & 15;
  int row0 = (lane >> 4) << 3;

  const _Float16* WT  = (mat == 0) ? WqT : (mat == 1) ? WkT : WvT;
  const float*    bia = (mat == 0) ? bq  : (mat == 1) ? bk  : bv;

  const _Float16* A  = enh + ((size_t)(t * N_) + nt * 16) * EP_;
  const _Float16* Bt = WT + ((size_t)(t * H_) + ct * 16) * EP_;
  v8f c = {};
#pragma unroll
  for (int kc = 0; kc < 3; ++kc)
    c = wmma_f16(load_a_tile(A + kc * 32, EP_), load_b_tile(Bt + kc * 32, EP_), c);

  int ocol = ct * 16 + col;
  float bb = bia[t * H_ + ocol];
  if (mat < 2) {
    _Float16* dst = ((mat == 0) ? q16 : k16) + ((size_t)(t * N_) + nt * 16 + row0) * H_ + ocol;
#pragma unroll
    for (int rr = 0; rr < 8; ++rr) dst[(size_t)rr * H_] = (_Float16)(c[rr] + bb);
  } else {
    // V stored transposed per head: vT[(t*NH+h)*HD + d][n]  (contiguous in n)
    int hh = ocol >> 5, dd = ocol & 31;
    union { unsigned short us[8]; uint4 v4; } pk;
#pragma unroll
    for (int rr = 0; rr < 8; ++rr) {
      _Float16 hv = (_Float16)(c[rr] + bb);
      pk.us[rr] = __builtin_bit_cast(unsigned short, hv);
    }
    *(uint4*)(vT + ((size_t)((t * NH_ + hh) * HD_ + dd)) * N_ + nt * 16 + row0) = pk.v4;
  }
}

// ---------- flash attention: block = (t, 16-row tile), 4 waves = 4 heads ----------
// Adjacency mask is head-invariant: build a 16x2048 bitmask once per block in LDS
// (4x HBM traffic reduction vs per-head reads), then each head-wave consumes one
// ds_load word per row per 32-column iteration.
__global__ void attn_kernel(const _Float16* __restrict__ q16, const _Float16* __restrict__ k16,
                            const _Float16* __restrict__ vT, const float* __restrict__ adj,
                            const float* __restrict__ ts, float* __restrict__ mh) {
  __shared__ unsigned maskw[16 * 64];       // [row][word] : 2048 cols as 64 u32 words
  __shared__ _Float16 pbuf[4 * 16 * 32];    // per-wave P re-layout buffer

  int nt = blockIdx.x & 127;
  int t  = blockIdx.x >> 7;
  int h  = threadIdx.x >> 5;                // wave-in-block = head
  int lane = threadIdx.x & 31;
  int col  = lane & 15;
  int row0 = (lane >> 4) << 3;
  _Float16* pb = pbuf + h * (16 * 32);

  // ---- cooperative mask build (all 128 threads) ----
  {
    const float* adjt = adj + ((size_t)t * N_ + nt * 16) * N_;
    for (int wi = threadIdx.x; wi < 16 * 64; wi += 128) {
      int row = wi >> 6, wc = wi & 63;
      const float* ap = adjt + (size_t)row * N_ + wc * 32;
      unsigned bits = 0;
#pragma unroll 8
      for (int b = 0; b < 32; ++b) bits |= (ap[b] > 0.f) ? (1u << b) : 0u;
      int grow = nt * 16 + row;
      if ((grow >> 5) == wc) bits |= (1u << (grow & 31));   // diagonal (self-edge)
      maskw[wi] = bits;
    }
  }
  __syncthreads();

  const float invs = 0.17677669529663687f;  // 1/sqrt(HD)

  v16h a = load_a_tile(q16 + ((size_t)(t * N_) + nt * 16) * H_ + h * HD_, H_);

  int rowbase = nt * 16 + row0;
  float tsn[8];
#pragma unroll
  for (int rr = 0; rr < 8; ++rr) tsn[rr] = ts[t * N_ + rowbase + rr];

  v8f o0 = {}, o1 = {};
  float mi[8], li[8];
#pragma unroll
  for (int rr = 0; rr < 8; ++rr) { mi[rr] = -1e30f; li[rr] = 0.f; }

  const _Float16* vbase = vT + ((size_t)((t * NH_ + h) * HD_)) * N_;

  for (int it = 0; it < N_ / 32; ++it) {
    int m0 = it * 32;
    __builtin_prefetch(k16 + ((size_t)(t * N_) + m0 + 32) * H_, 0, 1);
    v16h b0 = load_b_tile(k16 + ((size_t)(t * N_) + m0) * H_ + h * HD_, H_);
    v16h b1 = load_b_tile(k16 + ((size_t)(t * N_) + m0 + 16) * H_ + h * HD_, H_);
    v8f z0 = {}, z1 = {};
    v8f s0 = wmma_f16(a, b0, z0);
    v8f s1 = wmma_f16(a, b1, z1);

    float tsm0 = ts[t * N_ + m0 + col];
    float tsm1 = ts[t * N_ + m0 + 16 + col];

#pragma unroll
    for (int rr = 0; rr < 8; ++rr) {
      unsigned mw = maskw[(row0 + rr) * 64 + it];   // bits [0..15]=cols m0+0..15, [16..31]=m0+16..31
      float d0 = __expf(-fabsf(tsn[rr] - tsm0) * 0.1f);
      float d1 = __expf(-fabsf(tsn[rr] - tsm1) * 0.1f);
      float v0 = s0[rr] * invs * d0;
      float v1 = s1[rr] * invs * d1;
      s0[rr] = ((mw >> col) & 1u)        ? v0 : -3.0e38f;
      s1[rr] = ((mw >> (16 + col)) & 1u) ? v1 : -3.0e38f;
    }
    // online softmax per row (16-lane groups match C-layout row striping)
#pragma unroll
    for (int rr = 0; rr < 8; ++rr) {
      float tmax = fmaxf(s0[rr], s1[rr]);
#pragma unroll
      for (int m = 8; m >= 1; m >>= 1) tmax = fmaxf(tmax, __shfl_xor(tmax, m, 32));
      float newm = fmaxf(mi[rr], tmax);
      float corr = __expf(mi[rr] - newm);
      float p0 = __expf(s0[rr] - newm);
      float p1 = __expf(s1[rr] - newm);
      float ps = p0 + p1;
#pragma unroll
      for (int m = 8; m >= 1; m >>= 1) ps += __shfl_xor(ps, m, 32);
      li[rr] = li[rr] * corr + ps;
      mi[rr] = newm;
      o0[rr] *= corr;
      o1[rr] *= corr;
      pb[(row0 + rr) * 32 + col]      = (_Float16)p0;
      pb[(row0 + rr) * 32 + 16 + col] = (_Float16)p1;
    }
    // LDS bounce: re-layout P [16n x 32m] as WMMA A operand, then PV
    v16h pa  = load_a_tile(pb, 32);
    v16h bv0 = load_b_tile(vbase + m0, N_);
    v16h bv1 = load_b_tile(vbase + (size_t)16 * N_ + m0, N_);
    o0 = wmma_f16(pa, bv0, o0);
    o1 = wmma_f16(pa, bv1, o1);
  }

  float* outp = mh + ((size_t)(t * N_) + rowbase) * H_ + h * HD_;
#pragma unroll
  for (int rr = 0; rr < 8; ++rr) {
    float inv = 1.0f / li[rr];
    outp[(size_t)rr * H_ + col]      = o0[rr] * inv;
    outp[(size_t)rr * H_ + 16 + col] = o1[rr] * inv;
  }
}

// ---------- LayerNorm (one wave per row of 128) ----------
__global__ void ln_kernel(const float* __restrict__ mh, const float* __restrict__ ln_g,
                          const float* __restrict__ ln_b, _Float16* __restrict__ mhn) {
  int wave = (blockIdx.x * blockDim.x + threadIdx.x) >> 5;  // t*N + n
  if (wave >= T_ * N_) return;
  int lane = threadIdx.x & 31;
  int t = wave / N_;
  const float* rowp = mh + (size_t)wave * H_;
  float x[4], s = 0.f, s2 = 0.f;
#pragma unroll
  for (int i = 0; i < 4; ++i) { x[i] = rowp[lane + 32 * i]; s += x[i]; s2 += x[i] * x[i]; }
#pragma unroll
  for (int m = 16; m >= 1; m >>= 1) { s += __shfl_xor(s, m, 32); s2 += __shfl_xor(s2, m, 32); }
  float mu   = s * (1.0f / H_);
  float var  = s2 * (1.0f / H_) - mu * mu;
  float rstd = rsqrtf(var + 1e-6f);
  _Float16* outp = mhn + (size_t)wave * H_;
#pragma unroll
  for (int i = 0; i < 4; ++i) {
    int c = lane + 32 * i;
    outp[c] = (_Float16)(((x[i] - mu) * rstd) * ln_g[t * H_ + c] + ln_b[t * H_ + c]);
  }
}

// ---------- FFN layer 1: [N,128]@[128,256] + gelu ----------
__global__ void ffn1_kernel(const _Float16* __restrict__ mhn, const _Float16* __restrict__ W1T,
                            const float* __restrict__ b1, _Float16* __restrict__ g1) {
  int wave = (blockIdx.x * blockDim.x + threadIdx.x) >> 5;
  int ct = wave & 15; wave >>= 4;
  int nt = wave & 127; wave >>= 7;
  int t = wave;
  if (t >= T_) return;
  int lane = threadIdx.x & 31, col = lane & 15, row0 = (lane >> 4) << 3;
  const _Float16* A  = mhn + ((size_t)(t * N_) + nt * 16) * H_;
  const _Float16* Bt = W1T + ((size_t)(t * F_) + ct * 16) * H_;
  v8f c = {};
#pragma unroll
  for (int kc = 0; kc < 4; ++kc)
    c = wmma_f16(load_a_tile(A + kc * 32, H_), load_b_tile(Bt + kc * 32, H_), c);
  int ocol = ct * 16 + col;
  float bb = b1[t * F_ + ocol];
  _Float16* dst = g1 + ((size_t)(t * N_) + nt * 16 + row0) * F_ + ocol;
#pragma unroll
  for (int rr = 0; rr < 8; ++rr) dst[(size_t)rr * F_] = (_Float16)gelu_tanh(c[rr] + bb);
}

// ---------- FFN layer 2: [N,256]@[256,128] ----------
__global__ void ffn2_kernel(const _Float16* __restrict__ g1, const _Float16* __restrict__ W2T,
                            const float* __restrict__ b2, _Float16* __restrict__ ff16) {
  int wave = (blockIdx.x * blockDim.x + threadIdx.x) >> 5;
  int ct = wave & 7; wave >>= 3;
  int nt = wave & 127; wave >>= 7;
  int t = wave;
  if (t >= T_) return;
  int lane = threadIdx.x & 31, col = lane & 15, row0 = (lane >> 4) << 3;
  const _Float16* A  = g1  + ((size_t)(t * N_) + nt * 16) * F_;
  const _Float16* Bt = W2T + ((size_t)(t * H_) + ct * 16) * F_;
  v8f c = {};
#pragma unroll
  for (int kc = 0; kc < 8; ++kc)
    c = wmma_f16(load_a_tile(A + kc * 32, F_), load_b_tile(Bt + kc * 32, F_), c);
  int ocol = ct * 16 + col;
  float bb = b2[t * H_ + ocol];
  _Float16* dst = ff16 + ((size_t)(t * N_) + nt * 16 + row0) * H_ + ocol;
#pragma unroll
  for (int rr = 0; rr < 8; ++rr) dst[(size_t)rr * H_] = (_Float16)(c[rr] + bb);
}

// ---------- init h ----------
__global__ void zero_h_kernel(float* __restrict__ h, _Float16* __restrict__ h16) {
  int i = blockIdx.x * blockDim.x + threadIdx.x;
  if (i < N_ * H_) { h[i] = 0.f; h16[i] = (_Float16)0.f; }
}

// ---------- GRU r/z gates (per timestep) ----------
__global__ void gru_rz_kernel(const _Float16* __restrict__ h16, const _Float16* __restrict__ ff16_t,
                              const _Float16* __restrict__ WrT, const _Float16* __restrict__ WzT,
                              const float* __restrict__ br, const float* __restrict__ bz,
                              _Float16* __restrict__ rh16, float* __restrict__ zbuf) {
  int wave = (blockIdx.x * blockDim.x + threadIdx.x) >> 5;
  int ct = wave & 7;
  int nt = wave >> 3;
  if (nt >= 128) return;
  int lane = threadIdx.x & 31, col = lane & 15, row0 = (lane >> 4) << 3;
  const _Float16* Br = WrT + (size_t)(ct * 16) * F_;
  const _Float16* Bz = WzT + (size_t)(ct * 16) * F_;
  v8f cr = {}, cz = {};
#pragma unroll
  for (int kc = 0; kc < 8; ++kc) {
    const _Float16* A = (kc < 4) ? (h16 + (size_t)(nt * 16) * H_ + kc * 32)
                                 : (ff16_t + (size_t)(nt * 16) * H_ + (kc - 4) * 32);
    v16h a = load_a_tile(A, H_);
    cr = wmma_f16(a, load_b_tile(Br + kc * 32, F_), cr);
    cz = wmma_f16(a, load_b_tile(Bz + kc * 32, F_), cz);
  }
  int ocol = ct * 16 + col;
  float brv = br[ocol], bzv = bz[ocol];
#pragma unroll
  for (int rr = 0; rr < 8; ++rr) {
    int row = nt * 16 + row0 + rr;
    float rg = 1.0f / (1.0f + __expf(-(cr[rr] + brv)));
    float zg = 1.0f / (1.0f + __expf(-(cz[rr] + bzv)));
    zbuf[(size_t)row * H_ + ocol] = zg;
    rh16[(size_t)row * H_ + ocol] = (_Float16)(rg * (float)h16[(size_t)row * H_ + ocol]);
  }
}

// ---------- GRU candidate + state update (per timestep) ----------
__global__ void gru_n_kernel(const _Float16* __restrict__ rh16, const _Float16* __restrict__ ff16_t,
                             const _Float16* __restrict__ WnT, const float* __restrict__ bn,
                             const float* __restrict__ zbuf, float* __restrict__ h,
                             _Float16* __restrict__ h16) {
  int wave = (blockIdx.x * blockDim.x + threadIdx.x) >> 5;
  int ct = wave & 7;
  int nt = wave >> 3;
  if (nt >= 128) return;
  int lane = threadIdx.x & 31, col = lane & 15, row0 = (lane >> 4) << 3;
  const _Float16* Bn = WnT + (size_t)(ct * 16) * F_;
  v8f cn = {};
#pragma unroll
  for (int kc = 0; kc < 8; ++kc) {
    const _Float16* A = (kc < 4) ? (rh16 + (size_t)(nt * 16) * H_ + kc * 32)
                                 : (ff16_t + (size_t)(nt * 16) * H_ + (kc - 4) * 32);
    cn = wmma_f16(load_a_tile(A, H_), load_b_tile(Bn + kc * 32, F_), cn);
  }
  int ocol = ct * 16 + col;
  float bnv = bn[ocol];
#pragma unroll
  for (int rr = 0; rr < 8; ++rr) {
    int row = nt * 16 + row0 + rr;
    float nv = tanhf(cn[rr] + bnv);
    float zz = zbuf[(size_t)row * H_ + ocol];
    float hv = h[(size_t)row * H_ + ocol];
    float hn = (1.0f - zz) * nv + zz * hv;
    h[(size_t)row * H_ + ocol]   = hn;
    h16[(size_t)row * H_ + ocol] = (_Float16)hn;
  }
}

// ---------- final projection: [N,128]@[128,64] ----------
__global__ void out_kernel(const _Float16* __restrict__ h16, const _Float16* __restrict__ WoutT,
                           const float* __restrict__ bout, float* __restrict__ out) {
  int wave = (blockIdx.x * blockDim.x + threadIdx.x) >> 5;
  int ct = wave & 3;
  int nt = wave >> 2;
  if (nt >= 128) return;
  int lane = threadIdx.x & 31, col = lane & 15, row0 = (lane >> 4) << 3;
  const _Float16* A  = h16 + (size_t)(nt * 16) * H_;
  const _Float16* Bt = WoutT + (size_t)(ct * 16) * H_;
  v8f c = {};
#pragma unroll
  for (int kc = 0; kc < 4; ++kc)
    c = wmma_f16(load_a_tile(A + kc * 32, H_), load_b_tile(Bt + kc * 32, H_), c);
  int ocol = ct * 16 + col;
  float bb = bout[ocol];
#pragma unroll
  for (int rr = 0; rr < 8; ++rr)
    out[(size_t)(nt * 16 + row0 + rr) * OUT_ + ocol] = c[rr] + bb;
}

extern "C" void kernel_launch(void* const* d_in, const int* in_sizes, int n_in,
                              void* d_out, int out_size, void* d_ws, size_t ws_size,
                              hipStream_t stream) {
  const float* nodes = (const float*)d_in[0];
  const float* adj   = (const float*)d_in[1];
  const float* ts    = (const float*)d_in[2];
  const float* tf    = (const float*)d_in[4];
  const float* Wq = (const float*)d_in[5];  const float* bq = (const float*)d_in[6];
  const float* Wk = (const float*)d_in[7];  const float* bk = (const float*)d_in[8];
  const float* Wv = (const float*)d_in[9];  const float* bv = (const float*)d_in[10];
  const float* ln_g = (const float*)d_in[11]; const float* ln_b = (const float*)d_in[12];
  const float* W1 = (const float*)d_in[13]; const float* b1 = (const float*)d_in[14];
  const float* W2 = (const float*)d_in[15]; const float* b2 = (const float*)d_in[16];
  const float* Wr = (const float*)d_in[17]; const float* br = (const float*)d_in[18];
  const float* Wz = (const float*)d_in[19]; const float* bz = (const float*)d_in[20];
  const float* Wn = (const float*)d_in[21]; const float* bn = (const float*)d_in[22];
  const float* Wout = (const float*)d_in[23]; const float* bout = (const float*)d_in[24];
  float* out = (float*)d_out;

  char* ws = (char*)d_ws;
  size_t off = 0;
  auto alloc = [&](size_t bytes) -> void* {
    void* p = ws + off;
    off = (off + bytes + 255) & ~(size_t)255;
    return p;
  };
  _Float16* enh   = (_Float16*)alloc((size_t)T_ * N_ * EP_ * 2);
  _Float16* WqT   = (_Float16*)alloc((size_t)T_ * H_ * EP_ * 2);
  _Float16* WkT   = (_Float16*)alloc((size_t)T_ * H_ * EP_ * 2);
  _Float16* WvT   = (_Float16*)alloc((size_t)T_ * H_ * EP_ * 2);
  _Float16* W1T   = (_Float16*)alloc((size_t)T_ * F_ * H_ * 2);
  _Float16* W2T   = (_Float16*)alloc((size_t)T_ * H_ * F_ * 2);
  _Float16* WrT   = (_Float16*)alloc((size_t)T_ * H_ * F_ * 2);
  _Float16* WzT   = (_Float16*)alloc((size_t)T_ * H_ * F_ * 2);
  _Float16* WnT   = (_Float16*)alloc((size_t)T_ * H_ * F_ * 2);
  _Float16* WoutT = (_Float16*)alloc((size_t)OUT_ * H_ * 2);
  _Float16* q16   = (_Float16*)alloc((size_t)T_ * N_ * H_ * 2);
  _Float16* k16   = (_Float16*)alloc((size_t)T_ * N_ * H_ * 2);
  _Float16* vT    = (_Float16*)alloc((size_t)T_ * N_ * H_ * 2);
  float*    mh    = (float*)   alloc((size_t)T_ * N_ * H_ * 4);
  _Float16* mhn   = (_Float16*)alloc((size_t)T_ * N_ * H_ * 2);
  _Float16* g1    = (_Float16*)alloc((size_t)T_ * N_ * F_ * 2);
  _Float16* ff16  = (_Float16*)alloc((size_t)T_ * N_ * H_ * 2);
  float*    h     = (float*)   alloc((size_t)N_ * H_ * 4);
  _Float16* h16   = (_Float16*)alloc((size_t)N_ * H_ * 2);
  _Float16* rh16  = (_Float16*)alloc((size_t)N_ * H_ * 2);
  float*    zbuf  = (float*)   alloc((size_t)N_ * H_ * 4);
  (void)ws_size; (void)in_sizes; (void)n_in; (void)out_size;

  // 1) enhanced features + weight re-layouts
  prep_enh_kernel<<<(T_ * N_ + 255) / 256, 256, 0, stream>>>(nodes, ts, tf, enh);
  transpose_pad_kernel<<<256, 256, 0, stream>>>(Wq, WqT, T_, 80, H_, EP_);
  transpose_pad_kernel<<<256, 256, 0, stream>>>(Wk, WkT, T_, 80, H_, EP_);
  transpose_pad_kernel<<<256, 256, 0, stream>>>(Wv, WvT, T_, 80, H_, EP_);
  transpose_pad_kernel<<<256, 256, 0, stream>>>(W1, W1T, T_, H_, F_, H_);
  transpose_pad_kernel<<<256, 256, 0, stream>>>(W2, W2T, T_, F_, H_, F_);
  transpose_pad_kernel<<<256, 256, 0, stream>>>(Wr, WrT, T_, F_, H_, F_);
  transpose_pad_kernel<<<256, 256, 0, stream>>>(Wz, WzT, T_, F_, H_, F_);
  transpose_pad_kernel<<<256, 256, 0, stream>>>(Wn, WnT, T_, F_, H_, F_);
  transpose_pad_kernel<<<64, 256, 0, stream>>>(Wout, WoutT, 1, H_, OUT_, H_);

  // 2) QKV projections (24576 tile-waves, 8 waves/block)
  qkv_kernel<<<3072, 256, 0, stream>>>(enh, WqT, WkT, WvT, bq, bk, bv, q16, k16, vT);

  // 3) flash attention: 1024 blocks = (t, row-tile); 4 waves/block = 4 heads
  attn_kernel<<<T_ * 128, 128, 0, stream>>>(q16, k16, vT, adj, ts, mh);

  // 4) LayerNorm + FFN
  ln_kernel<<<(T_ * N_) / 8, 256, 0, stream>>>(mh, ln_g, ln_b, mhn);
  ffn1_kernel<<<(T_ * 128 * 16) / 8, 256, 0, stream>>>(mhn, W1T, b1, g1);
  ffn2_kernel<<<(T_ * 128 * 8) / 8, 256, 0, stream>>>(g1, W2T, b2, ff16);

  // 5) sequential GRU over T (only part with a true dependency chain)
  zero_h_kernel<<<(N_ * H_ + 255) / 256, 256, 0, stream>>>(h, h16);
  for (int t = 0; t < T_; ++t) {
    gru_rz_kernel<<<128, 256, 0, stream>>>(h16, ff16 + (size_t)t * N_ * H_,
                                           WrT + (size_t)t * H_ * F_, WzT + (size_t)t * H_ * F_,
                                           br + t * H_, bz + t * H_, rh16, zbuf);
    gru_n_kernel<<<128, 256, 0, stream>>>(rh16, ff16 + (size_t)t * N_ * H_,
                                          WnT + (size_t)t * H_ * F_, bn + t * H_,
                                          zbuf, h, h16);
  }

  // 6) output projection
  out_kernel<<<64, 256, 0, stream>>>(h16, WoutT, bout, out);
}
